// RKdAngle_53850299957760
// MI455X (gfx1250) — compile-verified, matrix operand order checked by reading, not run on hardware
//
#include <hip/hip_runtime.h>
#include <hip/hip_bf16.h>

#define SS   384
#define BB   4
#define DD   64
#define SPAD 8
#define LSTR (DD + SPAD)          // 72 halves = 144 B row pitch (bank-spread)
#define NB   12                   // 384/32 macro-tiles (32x32)
#define NUB  (NB * (NB + 1) / 2)  // 78 upper-triangle macro-tiles
#define NWAVE 8

typedef _Float16 v4h  __attribute__((ext_vector_type(4)));
typedef _Float16 v8h  __attribute__((ext_vector_type(8)));
typedef _Float16 v16h __attribute__((ext_vector_type(16)));
typedef float    v8f  __attribute__((ext_vector_type(8)));

// Load one A/B fragment (16x32 f16, K-slice kb=0 or 32) from an LDS matrix
// stored row-major [SS][LSTR]. Documented 16-bit A layout:
//   lane<16 : row = r0+lane,    K = {kb..kb+7, kb+16..kb+23}
//   lane>=16: row = r0+lane-16, K = {kb+8..kb+15, kb+24..kb+31}
// Two 16-byte ds_load_b128 per fragment.
__device__ __forceinline__ v16h load_frag(const _Float16* lds, int r0, int kb, int lane) {
    const _Float16* p = lds + (r0 + (lane & 15)) * LSTR + kb + ((lane >> 4) << 3);
    v8h lo = *reinterpret_cast<const v8h*>(p);
    v8h hi = *reinterpret_cast<const v8h*>(p + 16);
    return __builtin_shufflevector(lo, hi, 0,1,2,3,4,5,6,7,8,9,10,11,12,13,14,15);
}

// Build one normalized-difference row: lds_row[0..63] = f16(normalize(x[j]-x[i]))
// x layout is the reference's [S, B, D] so x[b, j, d] = in[(j*B+b)*D + d].
__device__ __forceinline__ void make_row(const float* __restrict__ xin,
                                         int b, int i, int j, _Float16* ldsrow) {
    const float4* xj = reinterpret_cast<const float4*>(xin + ((size_t)j * BB + b) * DD);
    const float4* xi = reinterpret_cast<const float4*>(xin + ((size_t)i * BB + b) * DD);
    float4 dv[16];
    float ss = 0.f;
#pragma unroll
    for (int c = 0; c < 16; ++c) {
        float4 a = xj[c], w = xi[c];
        float dx = a.x - w.x, dy = a.y - w.y, dz = a.z - w.z, dq = a.w - w.w;
        dv[c].x = dx; dv[c].y = dy; dv[c].z = dz; dv[c].w = dq;
        ss += dx * dx + dy * dy + dz * dz + dq * dq;
    }
    // j==i: ss==0 -> inv = 1/eps, but dv==0 so row stores exact zeros (matches ref)
    float inv = 1.0f / fmaxf(sqrtf(ss), 1e-12f);
#pragma unroll
    for (int c = 0; c < 16; ++c) {
        v4h h;
        h[0] = (_Float16)(dv[c].x * inv);
        h[1] = (_Float16)(dv[c].y * inv);
        h[2] = (_Float16)(dv[c].z * inv);
        h[3] = (_Float16)(dv[c].w * inv);
        *reinterpret_cast<v4h*>(ldsrow + 4 * c) = h;
    }
}

__global__ __launch_bounds__(256) void rkd_angle_kernel(const float* __restrict__ stu,
                                                        const float* __restrict__ tea,
                                                        float* __restrict__ partial) {
    __shared__ _Float16 lds_s[SS * LSTR];   // 54 KB
    __shared__ _Float16 lds_t[SS * LSTR];   // 54 KB

    const int tid  = threadIdx.x;
    const int bi   = blockIdx.x;            // i*B + b
    const int b    = bi & (BB - 1);
    const int i    = bi >> 2;

    // ---- Phase 1: normalized pairwise-difference rows -> LDS (f16) ----
    for (int j = tid; j < SS; j += 256) {
        make_row(stu, b, i, j, lds_s + j * LSTR);
        make_row(tea, b, i, j, lds_t + j * LSTR);
    }
    __syncthreads();

    // ---- Phase 2: 2x2 register-blocked WMMA Gram tiles + fused smooth-L1 ----
    const int wid  = tid >> 5;
    const int lane = tid & 31;
    float acc = 0.f;

    for (int t = wid; t < NUB; t += NWAVE) {
        // map linear upper-triangle index -> (bm, bn), bm <= bn (wave-uniform)
        int tt = t, bm = 0;
        while (tt >= NB - bm) { tt -= NB - bm; ++bm; }
        int bn = bm + tt;
        int m0 = bm << 5, n0 = bn << 5;

        // ---- student: load 8 fragments (16 ds_load_b128), do 8 WMMAs ----
        v16h sa[2][2], sb[2][2];               // [row-block][k-chunk]
#pragma unroll
        for (int r = 0; r < 2; ++r)
#pragma unroll
            for (int k = 0; k < 2; ++k) {
                sa[r][k] = load_frag(lds_s, m0 + (r << 4), k << 5, lane);
                sb[r][k] = load_frag(lds_s, n0 + (r << 4), k << 5, lane);
            }
        v8f cs[2][2];
#pragma unroll
        for (int r = 0; r < 2; ++r)
#pragma unroll
            for (int c = 0; c < 2; ++c) {
                v8f z = {};
                z = __builtin_amdgcn_wmma_f32_16x16x32_f16(false, sa[r][0], false, sb[c][0], (short)0, z, false, false);
                z = __builtin_amdgcn_wmma_f32_16x16x32_f16(false, sa[r][1], false, sb[c][1], (short)0, z, false, false);
                cs[r][c] = z;
            }

        // ---- teacher: same ----
        v16h ta[2][2], tb[2][2];
#pragma unroll
        for (int r = 0; r < 2; ++r)
#pragma unroll
            for (int k = 0; k < 2; ++k) {
                ta[r][k] = load_frag(lds_t, m0 + (r << 4), k << 5, lane);
                tb[r][k] = load_frag(lds_t, n0 + (r << 4), k << 5, lane);
            }
        v8f ct[2][2];
#pragma unroll
        for (int r = 0; r < 2; ++r)
#pragma unroll
            for (int c = 0; c < 2; ++c) {
                v8f z = {};
                z = __builtin_amdgcn_wmma_f32_16x16x32_f16(false, ta[r][0], false, tb[c][0], (short)0, z, false, false);
                z = __builtin_amdgcn_wmma_f32_16x16x32_f16(false, ta[r][1], false, tb[c][1], (short)0, z, false, false);
                ct[r][c] = z;
            }

        // off-diagonal macro-tiles counted twice (Gram symmetry); diagonal
        // macro-tiles already contain both (i,j) and (j,i) sub-tiles -> weight 1
        float w = (bm == bn) ? 1.0f : 2.0f;
        float tl = 0.f;
#pragma unroll
        for (int r = 0; r < 2; ++r)
#pragma unroll
            for (int c = 0; c < 2; ++c)
#pragma unroll
                for (int e = 0; e < 8; ++e) {
                    float df = cs[r][c][e] - ct[r][c][e];
                    float ad = fabsf(df);
                    tl += (ad < 1.0f) ? (0.5f * df * df) : (ad - 0.5f);
                }
        acc += w * tl;
    }

    // ---- Phase 3: workgroup reduction (reuse LDS), write partial ----
    __syncthreads();
    float* red = reinterpret_cast<float*>(lds_s);
    red[tid] = acc;
    __syncthreads();
#pragma unroll
    for (int off = 128; off > 0; off >>= 1) {
        if (tid < off) red[tid] += red[tid + off];
        __syncthreads();
    }
    if (tid == 0) partial[bi] = red[0];
}

__global__ __launch_bounds__(256) void rkd_reduce_kernel(const float* __restrict__ partial,
                                                         float* __restrict__ out) {
    __shared__ float red[256];
    float s = 0.f;
    for (int idx = threadIdx.x; idx < SS * BB; idx += 256) s += partial[idx];
    red[threadIdx.x] = s;
    __syncthreads();
#pragma unroll
    for (int off = 128; off > 0; off >>= 1) {
        if (threadIdx.x < off) red[threadIdx.x] += red[threadIdx.x + off];
        __syncthreads();
    }
    if (threadIdx.x == 0) {
        // mean over B*S^3 = 4*384^3 elements
        out[0] = red[0] * (1.0f / 226492416.0f);
    }
}

extern "C" void kernel_launch(void* const* d_in, const int* in_sizes, int n_in,
                              void* d_out, int out_size, void* d_ws, size_t ws_size,
                              hipStream_t stream) {
    const float* stu = (const float*)d_in[0];
    const float* tea = (const float*)d_in[1];
    float* partial   = (float*)d_ws;       // 1536 floats
    float* out       = (float*)d_out;

    rkd_angle_kernel<<<SS * BB, 256, 0, stream>>>(stu, tea, partial);
    rkd_reduce_kernel<<<1, 256, 0, stream>>>(partial, out);
}
